// LSTMClassifier_7945689498053
// MI455X (gfx1250) — compile-verified, matrix operand order checked by reading, not run on hardware
//
#include <hip/hip_runtime.h>
#include <cstdint>
#include <cstddef>

typedef __bf16        bf16x16 __attribute__((ext_vector_type(16)));
typedef float         f32x8   __attribute__((ext_vector_type(8)));
typedef float         f32x4   __attribute__((ext_vector_type(4)));
typedef unsigned int  u32x4   __attribute__((ext_vector_type(4)));
typedef int           v4i     __attribute__((vector_size(16)));   // builtin arg type

#define B_SZ   256
#define T_SZ   512
#define D_SZ   512
#define H_SZ   1024
#define KTOT   1536      // D + H (fused input+recurrent GEMM)
#define NGATE  4096      // 4*H
#define OUT_SZ 8

#define MT 32            // batch rows per workgroup
#define JT 64            // hidden columns per workgroup
#define QT 256           // gate columns per workgroup (4 gates * JT)
#define KC 32            // K chunk per WMMA pass
#define KP 40            // padded LDS row length in u16 (80 B, conflict-avoiding)
#define NCHUNK (KTOT / KC)

// ---- gfx1250 async global->LDS copy (ASYNCcnt path), with safe fallback ----
#if __has_builtin(__builtin_amdgcn_global_load_async_to_lds_b128)
#define HAVE_ASYNC_LDS 1
#endif

#if __has_builtin(__builtin_amdgcn_s_wait_asynccnt)
#define WAIT_ASYNC() __builtin_amdgcn_s_wait_asynccnt(0)
#else
#define WAIT_ASYNC() asm volatile("s_wait_asynccnt 0x0" ::: "memory")
#endif

__device__ __forceinline__ unsigned short f2bf(float f) {
    unsigned u = __float_as_uint(f);
    u += 0x7FFFu + ((u >> 16) & 1u);      // round-to-nearest-even
    return (unsigned short)(u >> 16);
}

union FragU { bf16x16 v; u32x4 u[2]; };

// ---------------------------------------------------------------------------
// Prep: pack [W_ih | W_hh] -> bf16, rows pre-swizzled into per-(jtile) tile
// order (q = gate*64 + j), and bias = b_ih + b_hh.  Runs once per launch.
// ---------------------------------------------------------------------------
__global__ __launch_bounds__(256) void prep_kernel(
    const float* __restrict__ W_ih, const float* __restrict__ W_hh,
    const float* __restrict__ b_ih, const float* __restrict__ b_hh,
    unsigned short* __restrict__ Bpack, float* __restrict__ bias)
{
    const int n     = blockIdx.x;        // packed row 0..4095
    const int jtile = n >> 8;
    const int q     = n & 255;
    const int gate  = q >> 6;
    const int j     = q & 63;
    const int g     = gate * H_SZ + jtile * JT + j;   // original gate row
    for (int k = threadIdx.x; k < KTOT; k += 256) {
        float w = (k < D_SZ) ? W_ih[(size_t)g * D_SZ + k]
                             : W_hh[(size_t)g * H_SZ + (k - D_SZ)];
        Bpack[(size_t)n * KTOT + k] = f2bf(w);
    }
    if (threadIdx.x == 0) bias[g] = b_ih[g] + b_hh[g];
}

// ---------------------------------------------------------------------------
// One fused LSTM timestep:
//   gates[32 x 256] = [x_t | h_in] (bf16) @ Bpack_tile^T  + bias
//   c,h update in epilogue (fp32, in-LDS) -> h_out, c_state
// ---------------------------------------------------------------------------
__global__ __launch_bounds__(256) void lstm_step_kernel(
    const float* __restrict__ x, const unsigned short* __restrict__ Bpack,
    const float* __restrict__ bias, const float* __restrict__ h_in,
    float* __restrict__ h_out, float* __restrict__ c_state, int t)
{
    // staging: A double-buffer (2*32*40*2 = 5120 B) + B double-buffer
    // (2*256*40*2 = 40960 B); gate scratch (32*256*4 = 32768 B) aliases it.
    __shared__ __align__(16) unsigned char smem[2 * MT * KP * 2 + 2 * QT * KP * 2];

    const int tid    = threadIdx.x;
    const int jtile  = blockIdx.x;          // 0..15
    const int m0     = blockIdx.y * MT;     // 0..224
    const int lane   = tid & 31;
    const int wave   = tid >> 5;            // 8 waves
    const int m_sub  = wave & 1;            // 2 x 16-row subtiles
    const int n_base = (wave >> 1) * 64;    // 4 x 64-col wave stripes

    auto As = [&](int buf) -> unsigned short* {
        return (unsigned short*)(smem + buf * (MT * KP * 2));
    };
    auto Bs = [&](int buf) -> unsigned short* {
        return (unsigned short*)(smem + 2 * (MT * KP * 2) + buf * (QT * KP * 2));
    };

    f32x8 acc[4];
#pragma unroll
    for (int i = 0; i < 4; ++i)
        acc[i] = (f32x8){0.f, 0.f, 0.f, 0.f, 0.f, 0.f, 0.f, 0.f};

    // --- stage one K-chunk into LDS ----------------------------------------
    auto stage = [&](int kc, int buf) {
        const int k0 = kc * KC;
        {   // A tile: 32 rows x 32 k ; fp32->bf16 in flight (2 KB)
            const int r  = tid >> 3;
            const int kq = (tid & 7) * 4;
            const int k  = k0 + kq;
            f32x4 v;
            if (k < D_SZ)
                v = *(const f32x4*)(x + ((size_t)(m0 + r) * T_SZ + t) * D_SZ + k);
            else
                v = *(const f32x4*)(h_in + (size_t)(m0 + r) * H_SZ + (k - D_SZ));
            unsigned short* dst = As(buf) + r * KP + kq;
            dst[0] = f2bf(v[0]); dst[1] = f2bf(v[1]);
            dst[2] = f2bf(v[2]); dst[3] = f2bf(v[3]);
        }
        {   // B tile: 256 rows x 32 k ; pure bf16 byte copy (16 KB)
            const unsigned short* src =
                Bpack + (size_t)(jtile * QT + tid) * KTOT + k0;
            unsigned short* dst = Bs(buf) + tid * KP;
#ifdef HAVE_ASYNC_LDS
            // gfx1250 async DMA into LDS, no VGPR round trip; inst_offset
            // applies to BOTH global and LDS addresses.
            v4i* gp = (v4i*)const_cast<unsigned short*>(src);
            v4i* lp = (v4i*)dst;
            __builtin_amdgcn_global_load_async_to_lds_b128(gp, lp, 0,  0);
            __builtin_amdgcn_global_load_async_to_lds_b128(gp, lp, 16, 0);
            __builtin_amdgcn_global_load_async_to_lds_b128(gp, lp, 32, 0);
            __builtin_amdgcn_global_load_async_to_lds_b128(gp, lp, 48, 0);
#else
            u32x4 b0 = ((const u32x4*)src)[0];
            u32x4 b1 = ((const u32x4*)src)[1];
            u32x4 b2 = ((const u32x4*)src)[2];
            u32x4 b3 = ((const u32x4*)src)[3];
            u32x4* d4 = (u32x4*)dst;
            d4[0] = b0; d4[1] = b1; d4[2] = b2; d4[3] = b3;
#endif
            if (kc + 2 < NCHUNK)                      // warm L2 for k+2
                __builtin_prefetch(src + 2 * KC, 0, 0);
        }
    };

    // --- WMMA pass over one K-chunk -----------------------------------------
    // Frag layout per ISA: lanes 0-15 hold K{0..7,16..23}, lanes 16-31 hold
    // K{8..15,24..31}; A lane = row M, B lane = col N.  All DS loads are
    // issued before the WMMA chain so only one dscnt wait is needed.
    auto compute = [&](int buf) {
        const int half = lane >> 4;
        const int ml   = lane & 15;
        FragU a;
        const unsigned short* ar = As(buf) + (m_sub * 16 + ml) * KP + 8 * half;
        a.u[0] = *(const u32x4*)(ar);
        a.u[1] = *(const u32x4*)(ar + 16);
        FragU b[4];
#pragma unroll
        for (int tn = 0; tn < 4; ++tn) {
            const unsigned short* br =
                Bs(buf) + (n_base + tn * 16 + ml) * KP + 8 * half;
            b[tn].u[0] = *(const u32x4*)(br);
            b[tn].u[1] = *(const u32x4*)(br + 16);
        }
#pragma unroll
        for (int tn = 0; tn < 4; ++tn)
            acc[tn] = __builtin_amdgcn_wmma_f32_16x16x32_bf16(
                false, a.v, false, b[tn].v, (short)0, acc[tn], false, false);
    };

    // --- main K loop, double-buffered; async copy of chunk k+1 overlaps
    //     the WMMAs of chunk k ------------------------------------------------
    stage(0, 0);
    int buf = 0;
    for (int kc = 0; kc < NCHUNK; ++kc) {
#ifdef HAVE_ASYNC_LDS
        WAIT_ASYNC();                 // my async writes for chunk kc landed
#endif
        __syncthreads();              // everyone's staging of kc visible
        if (kc + 1 < NCHUNK) stage(kc + 1, buf ^ 1);
        compute(buf);
        buf ^= 1;
    }
    __syncthreads();

    // --- spill accumulators (+bias) to LDS gate scratch ---------------------
    float* Gs = (float*)smem;                 // [MT][QT]
    {
        const int ml = lane & 15;
        const int mh = (lane >> 4) * 8;
#pragma unroll
        for (int tn = 0; tn < 4; ++tn) {
            const int n  = n_base + tn * 16 + ml;
            const float bn = bias[(n >> 6) * H_SZ + jtile * JT + (n & 63)];
#pragma unroll
            for (int r = 0; r < 8; ++r) {
                const int m = m_sub * 16 + mh + r;
                Gs[m * QT + n] = acc[tn][r] + bn;
            }
        }
    }
    __syncthreads();

    // --- fused LSTM cell update (torch gate order i,f,g,o) ------------------
    for (int e = tid; e < MT * JT; e += 256) {
        const int m = e >> 6;
        const int j = e & 63;
        float gi = Gs[m * QT + j];
        float gf = Gs[m * QT + 64 + j];
        float gg = Gs[m * QT + 128 + j];
        float go = Gs[m * QT + 192 + j];
        gi = 1.f / (1.f + __expf(-gi));
        gf = 1.f / (1.f + __expf(-gf));
        gg = tanhf(gg);
        go = 1.f / (1.f + __expf(-go));
        const size_t off = (size_t)(m0 + m) * H_SZ + jtile * JT + j;
        const float cn = gf * c_state[off] + gi * gg;
        c_state[off] = cn;
        h_out[off]   = go * tanhf(cn);
    }
}

// ---------------------------------------------------------------------------
// Final FC + sigmoid: out[256 x 8] = sigmoid(h @ W_fc^T + b_fc).  4 MFLOP.
// ---------------------------------------------------------------------------
__global__ __launch_bounds__(256) void fc_kernel(
    const float* __restrict__ h, const float* __restrict__ W_fc,
    const float* __restrict__ b_fc, float* __restrict__ out)
{
    const int b = threadIdx.x;     // one thread per batch row
    float s[OUT_SZ];
#pragma unroll
    for (int o = 0; o < OUT_SZ; ++o) s[o] = b_fc[o];
    for (int k = 0; k < H_SZ; k += 4) {
        const f32x4 hv = *(const f32x4*)(h + (size_t)b * H_SZ + k);
#pragma unroll
        for (int o = 0; o < OUT_SZ; ++o) {
            const f32x4 wv = *(const f32x4*)(W_fc + (size_t)o * H_SZ + k);
            s[o] += hv[0] * wv[0] + hv[1] * wv[1] + hv[2] * wv[2] + hv[3] * wv[3];
        }
    }
#pragma unroll
    for (int o = 0; o < OUT_SZ; ++o)
        out[(size_t)b * OUT_SZ + o] = 1.f / (1.f + __expf(-s[o]));
}

// ---------------------------------------------------------------------------
extern "C" void kernel_launch(void* const* d_in, const int* in_sizes, int n_in,
                              void* d_out, int out_size, void* d_ws, size_t ws_size,
                              hipStream_t stream)
{
    const float* x    = (const float*)d_in[0];
    const float* W_ih = (const float*)d_in[1];
    const float* W_hh = (const float*)d_in[2];
    const float* b_ih = (const float*)d_in[3];
    const float* b_hh = (const float*)d_in[4];
    const float* W_fc = (const float*)d_in[5];
    const float* b_fc = (const float*)d_in[6];

    char* ws = (char*)d_ws;
    size_t off = 0;
    unsigned short* Bpack = (unsigned short*)(ws + off); off += (size_t)NGATE * KTOT * 2; // 12.58 MB
    float* bias = (float*)(ws + off); off += (size_t)NGATE * 4;
    float* h0   = (float*)(ws + off); off += (size_t)B_SZ * H_SZ * 4;
    float* h1   = (float*)(ws + off); off += (size_t)B_SZ * H_SZ * 4;
    float* cs   = (float*)(ws + off); off += (size_t)B_SZ * H_SZ * 4;
    if (off > ws_size) return;   // ~15.7 MB required

    (void)hipMemsetAsync(h0, 0, (size_t)B_SZ * H_SZ * 4, stream);
    (void)hipMemsetAsync(cs, 0, (size_t)B_SZ * H_SZ * 4, stream);

    prep_kernel<<<dim3(NGATE), dim3(256), 0, stream>>>(W_ih, W_hh, b_ih, b_hh,
                                                       Bpack, bias);

    float* hin = h0;
    float* hout = h1;
    for (int t = 0; t < T_SZ; ++t) {
        lstm_step_kernel<<<dim3(H_SZ / JT, B_SZ / MT), dim3(256), 0, stream>>>(
            x, Bpack, bias, hin, hout, cs, t);
        float* tmp = hin; hin = hout; hout = tmp;
    }
    // T even -> final h is back in h0 (== hin after the loop)
    fc_kernel<<<dim3(1), dim3(256), 0, stream>>>(hin, W_fc, b_fc, (float*)d_out);
}
